// NT_Xent_84559316124184
// MI455X (gfx1250) — compile-verified, matrix operand order checked by reading, not run on hardware
//
#include <hip/hip_runtime.h>
#include <hip/hip_bf16.h>

// ---------------------------------------------------------------------------
// NT-Xent (SimCLR) loss, N = 8192 rows (z_i || z_j), D = 128, TEMP = 0.5.
//
// loss = (1/N) * sum_i [ 2 + log( exp(pos_i-2)
//                                 + sum_{all k} exp(G[i,k]*2 - 2)
//                                 - exp(G[i,i]*2-2) - exp(G[i,p(i)]*2-2) )
//                        - pos_i ],   p(i) = (i+4096) mod 8192
//
// G = Zn Zn^T via f16 WMMA / f32 accumulate; fixed logsumexp shift of 2.0
// (cosine bound) removes the max pass. Excluded diagonal/partner terms are
// subtracted afterwards so the fused GEMM+exp inner loop is branchless.
// exp2 args are bounded in [-5.8, 0] -> raw v_exp_f32 (no denorm fixup).
// ---------------------------------------------------------------------------

typedef __attribute__((ext_vector_type(16))) _Float16 v16h;
typedef __attribute__((ext_vector_type(8)))  _Float16 v8h;
typedef __attribute__((ext_vector_type(4)))  _Float16 v4h;
typedef __attribute__((ext_vector_type(8)))  float    v8f;

constexpr int NROWS = 8192;
constexpr int HALF  = 4096;
constexpr int DIM   = 128;
constexpr int COLS_PER_STRIP = 1024;              // 64 column tiles of 16
constexpr int NSTRIPS = NROWS / COLS_PER_STRIP;   // 8

// exp(x*2 - 2) == exp2(x*K - K), K = 2*log2(e)
#define K2E 2.885390081777926814f

// ---------------------------------------------------------------------------
// Phase 1: normalize rows, emit f16 Zn, fp32 pos_i = ||zn_i||^2 / TEMP,
// and zero the per-row exp-sum accumulator. One wave (32 lanes) per row.
// ---------------------------------------------------------------------------
__global__ void ntx_normalize(const float* __restrict__ zi,
                              const float* __restrict__ zj,
                              _Float16* __restrict__ Zh,
                              float* __restrict__ pos,
                              float* __restrict__ rowsum) {
    const int wave = threadIdx.x >> 5;
    const int lane = threadIdx.x & 31;
    const int row  = blockIdx.x * 8 + wave;          // 1024 blocks * 8 waves

    const float* src = (row < HALF) ? (zi + (size_t)row * DIM)
                                    : (zj + (size_t)(row - HALF) * DIM);
    float4 x = *reinterpret_cast<const float4*>(src + lane * 4);
    float s = x.x * x.x + x.y * x.y + x.z * x.z + x.w * x.w;
    #pragma unroll
    for (int m = 1; m < 32; m <<= 1) s += __shfl_xor(s, m, 32);

    const float inv = 1.0f / fmaxf(sqrtf(s), 1e-8f);
    // pos_i = ||zn||^2 / TEMP  (TEMP = 0.5 -> *2)
    const float p = s * inv * inv * 2.0f;

    v4h hv;
    hv.x = (_Float16)(x.x * inv);
    hv.y = (_Float16)(x.y * inv);
    hv.z = (_Float16)(x.z * inv);
    hv.w = (_Float16)(x.w * inv);
    *reinterpret_cast<v4h*>(Zh + (size_t)row * DIM + lane * 4) = hv;

    if (lane == 0) {
        pos[row]    = p;
        rowsum[row] = 0.0f;
    }
}

// ---------------------------------------------------------------------------
// f16 A/B fragment load for V_WMMA_F32_16X16X32_F16.
// ISA 16-bit A layout (B mirrors with N in place of M): lane half h = lane>>4,
// element e: K = kOff + e + 8h for e<8, K = kOff + e + 8 + 8h for e>=8
// -> two contiguous 16-byte chunks per lane.
// ---------------------------------------------------------------------------
__device__ __forceinline__ v16h load_frag(const _Float16* __restrict__ Z,
                                          int row, int kOff, int h) {
    const _Float16* p = Z + (size_t)row * DIM + kOff + 8 * h;
    v8h lo = *reinterpret_cast<const v8h*>(p);
    v8h hi = *reinterpret_cast<const v8h*>(p + 16);
    v16h r;
    #pragma unroll
    for (int i = 0; i < 8; ++i) { r[i] = lo[i]; r[i + 8] = hi[i]; }
    return r;
}

// ---------------------------------------------------------------------------
// Phase 2: fused Gram-GEMM + unconditional exp accumulation (branchless).
// Each wave: one 16-row tile x one 1024-column strip (64 tiles x 4 WMMA).
// C layout: lane n = lane&15, half h = lane>>4; C[v] -> (m = v + 8h, n).
// ---------------------------------------------------------------------------
__global__ void ntx_gram_lse(const _Float16* __restrict__ Zh,
                             float* __restrict__ rowsum) {
    const int wave = threadIdx.x >> 5;
    const int lane = threadIdx.x & 31;
    const int unit = blockIdx.x * 8 + wave;          // 0..4095
    const int rowTile = unit >> 3;                   // 0..511
    const int strip   = unit & (NSTRIPS - 1);        // 0..7
    const int rowBase = rowTile * 16;
    const int h   = lane >> 4;
    const int lan = lane & 15;

    // A fragments: 16 rows x K=128, resident in registers for the whole strip.
    v16h a0 = load_frag(Zh, rowBase + lan,  0, h);
    v16h a1 = load_frag(Zh, rowBase + lan, 32, h);
    v16h a2 = load_frag(Zh, rowBase + lan, 64, h);
    v16h a3 = load_frag(Zh, rowBase + lan, 96, h);

    float racc[8];
    #pragma unroll
    for (int v = 0; v < 8; ++v) racc[v] = 0.0f;

    const int colBase0 = strip * COLS_PER_STRIP;
    for (int ct = 0; ct < COLS_PER_STRIP / 16; ++ct) {
        const int brow = colBase0 + ct * 16 + lan;
        v16h b0 = load_frag(Zh, brow,  0, h);
        v16h b1 = load_frag(Zh, brow, 32, h);
        v16h b2 = load_frag(Zh, brow, 64, h);
        v16h b3 = load_frag(Zh, brow, 96, h);

        v8f c = {};
        c = __builtin_amdgcn_wmma_f32_16x16x32_f16(false, a0, false, b0,
                                                   (short)0, c, false, false);
        c = __builtin_amdgcn_wmma_f32_16x16x32_f16(false, a1, false, b1,
                                                   (short)0, c, false, false);
        c = __builtin_amdgcn_wmma_f32_16x16x32_f16(false, a2, false, b2,
                                                   (short)0, c, false, false);
        c = __builtin_amdgcn_wmma_f32_16x16x32_f16(false, a3, false, b3,
                                                   (short)0, c, false, false);

        // exp(sim - 2) = exp2(c*K2E - K2E); arg in [-5.8, 0] -> raw v_exp_f32
        #pragma unroll
        for (int v = 0; v < 8; ++v)
            racc[v] += __builtin_amdgcn_exp2f(fmaf(c[v], K2E, -K2E));
    }

    // reduce each row accumulator across its 16-lane half, then atomic add
    #pragma unroll
    for (int v = 0; v < 8; ++v) {
        float s = racc[v];
        #pragma unroll
        for (int m = 1; m < 16; m <<= 1) s += __shfl_xor(s, m, 32);
        if (lan == 0) atomicAdd(&rowsum[rowBase + v + 8 * h], s);
    }
}

// ---------------------------------------------------------------------------
// Phase 2b: correction terms, one wave per row:
//   corr[i] = exp(G[i,i]*2-2) + exp(G[i,p(i)]*2-2), from the same f16 Zh.
// ---------------------------------------------------------------------------
__global__ void ntx_corr(const _Float16* __restrict__ Zh,
                         float* __restrict__ corr) {
    const int wave = threadIdx.x >> 5;
    const int lane = threadIdx.x & 31;
    const int row  = blockIdx.x * 8 + wave;
    const int prow = (row + HALF) & (NROWS - 1);

    v4h a = *reinterpret_cast<const v4h*>(Zh + (size_t)row  * DIM + lane * 4);
    v4h b = *reinterpret_cast<const v4h*>(Zh + (size_t)prow * DIM + lane * 4);
    float dd = 0.0f, cc = 0.0f;
    #pragma unroll
    for (int i = 0; i < 4; ++i) {
        float ai = (float)a[i];
        float bi = (float)b[i];
        dd = fmaf(ai, ai, dd);
        cc = fmaf(ai, bi, cc);
    }
    #pragma unroll
    for (int m = 1; m < 32; m <<= 1) {
        dd += __shfl_xor(dd, m, 32);
        cc += __shfl_xor(cc, m, 32);
    }
    if (lane == 0)
        corr[row] = __builtin_amdgcn_exp2f(fmaf(dd, K2E, -K2E)) +
                    __builtin_amdgcn_exp2f(fmaf(cc, K2E, -K2E));
}

// ---------------------------------------------------------------------------
// Phase 3: single-block finalize.
// loss_i = 2 + log(rowsum - corr + exp(pos-2)) - pos;  out = mean(loss_i).
// ---------------------------------------------------------------------------
__global__ void ntx_finalize(const float* __restrict__ rowsum,
                             const float* __restrict__ pos,
                             const float* __restrict__ corr,
                             float* __restrict__ out) {
    constexpr float LOG2E  = 1.442695040888963407f;   // log2(e)
    constexpr float LN2    = 0.693147180559945309f;   // ln(2)
    __shared__ float red[256];
    float local = 0.0f;
    for (int i = threadIdx.x; i < NROWS; i += 256) {
        const float p = pos[i];
        // exp(p-2) = exp2(p*log2e - 2*log2e); p ~ 2 -> arg ~ 0 (safe)
        const float ep = __builtin_amdgcn_exp2f(fmaf(p, LOG2E, -2.0f * LOG2E));
        const float s  = rowsum[i] - corr[i] + ep;
        // log(s) = log2(s) * ln2; s ~ O(1e3) (safe, normal range)
        local += 2.0f + __builtin_amdgcn_logf(s) * LN2 - p;
    }
    red[threadIdx.x] = local;
    __syncthreads();
    #pragma unroll
    for (int st = 128; st > 0; st >>= 1) {
        if (threadIdx.x < st) red[threadIdx.x] += red[threadIdx.x + st];
        __syncthreads();
    }
    if (threadIdx.x == 0) out[0] = red[0] / (float)NROWS;
}

// ---------------------------------------------------------------------------
extern "C" void kernel_launch(void* const* d_in, const int* in_sizes, int n_in,
                              void* d_out, int out_size, void* d_ws, size_t ws_size,
                              hipStream_t stream) {
    const float* zi = (const float*)d_in[0];
    const float* zj = (const float*)d_in[1];
    float* out = (float*)d_out;

    _Float16* Zh     = (_Float16*)d_ws;                            // 2 MB
    float*    rowsum = (float*)((char*)d_ws + (size_t)NROWS * DIM * sizeof(_Float16));
    float*    pos    = rowsum + NROWS;                             // +32 KB
    float*    corr   = pos + NROWS;                                // +32 KB

    ntx_normalize<<<NROWS / 8, 256, 0, stream>>>(zi, zj, Zh, pos, rowsum);
    ntx_corr<<<NROWS / 8, 256, 0, stream>>>(Zh, corr);
    ntx_gram_lse<<<(NROWS / 16) * NSTRIPS / 8, 256, 0, stream>>>(Zh, rowsum);
    ntx_finalize<<<1, 256, 0, stream>>>(rowsum, pos, corr, out);
}
